// ASG_26388279067346
// MI455X (gfx1250) — matrix operand order repaired and checked by reading, not compiled, not run on hardware
//
#include <hip/hip_runtime.h>
#include <hip/hip_bf16.h>

// ---- Problem constants (from reference) ----
#define T_STEPS 2000
#define BATCH   32
#define NC      514     // NUM_CLASSES + NUM_REPLABELS + 1
#define KP      544     // K padded to 17 tiles of 32
#define NRE     544     // E rows padded to 34 column tiles of 16
#define KTILES  17
#define LT      50
#define LL      101     // 2*LT+1
#define NEGV    (-1.0e30f)

typedef _Float16 v16h __attribute__((ext_vector_type(16)));
typedef _Float16 h8   __attribute__((ext_vector_type(8)));
typedef float    v8f  __attribute__((ext_vector_type(8)));

// ---------------------------------------------------------------------------
// Kernel 1: E[i][k] = exp(transmat[i][k]) = exp(trans[1+i][k]), f16.
// Row-major, row stride KP, padded to NRE rows (zeros). L2-resident (0.59 MB).
// ---------------------------------------------------------------------------
__global__ void asg_build_expT(const float* __restrict__ trans,
                               _Float16* __restrict__ Eg) {
  int idx = blockIdx.x * blockDim.x + threadIdx.x;
  if (idx >= NRE * KP) return;
  int i = idx / KP;
  int k = idx - i * KP;
  float v = (i < NC && k < NC) ? __expf(trans[(1 + i) * NC + k]) : 0.0f;
  Eg[idx] = (_Float16)v;
}

// ---------------------------------------------------------------------------
// Kernel 2: FCC scan. One persistent workgroup (512 threads = 16 waves).
// Per step: out[b,i] = sum_k p[b,k]*E[i,k] via v_wmma_f32_16x16x32_f16,
// then alpha_new[b,i] = inputs[b,t,i] + amax[b] + log(out[b,i]),
// then per-batch max + p = exp(alpha - amax) back into LDS (f16).
// Loop nest: kt outer, wave's 3 column-tiles inner; all 6 accumulators live.
// ---------------------------------------------------------------------------
__global__ __launch_bounds__(512, 1)
void asg_fcc_kernel(const float* __restrict__ inp,
                    const float* __restrict__ trans,
                    const _Float16* __restrict__ Eg,
                    float* __restrict__ fcc) {
  __shared__ _Float16 pbuf[BATCH][KP];     // 34816 B : p in f16 (A operand)
  __shared__ float alpha_s[BATCH][NRE];    // 69632 B : current alpha (NC used)
  __shared__ float red_s[BATCH][16];       //  2048 B
  __shared__ float amax_s[BATCH];          //   128 B

  const int tid  = threadIdx.x;
  const int wave = tid >> 5;
  const int lane = tid & 31;
  const int rb   = tid >> 4;               // reduction row (batch), 0..31
  const int g    = tid & 15;               // reduction group lane, 0..15

  const int mrow = lane & 15;              // A: M = lane%16 ; C/D: N = lane%16
  const int ksel = (lane & 16) ? 8 : 0;    // A half-lane K chunk select
  const int kbse = (lane & 16) ? 8 : 0;    // B half-lane K base
  const int rofs = (lane & 16) ? 8 : 0;    // C layout: M = r + 8*(lane/16)

  int col[3];
#pragma unroll
  for (int j = 0; j < 3; ++j) col[j] = (wave + 16 * j) * 16 + mrow; // < 768

  for (int t = 0; t < T_STEPS; ++t) {
    if (t == 0) {
      // alpha0 = inputs[:,0] + trans[0]
      for (int e = tid; e < BATCH * NC; e += 512) {
        int b = e / NC, i = e - b * NC;
        alpha_s[b][i] = inp[(size_t)b * T_STEPS * NC + i] + trans[i];
      }
    } else {
      // GEMM: p(32 x KP, f16 in LDS)  x  E^T  ->  32 x 544, f32 accum
      v8f acc[3][2];
#pragma unroll
      for (int j = 0; j < 3; ++j) { acc[j][0] = {}; acc[j][1] = {}; }

#pragma unroll 1
      for (int kt = 0; kt < KTILES; ++kt) {
        // A fragments (ISA 16-bit A layout): chunks {0-7,16-23}/{8-15,24-31}
        const _Float16* pr0 = &pbuf[mrow][kt * 32 + ksel];
        h8 a0lo = *(const h8*)pr0;
        h8 a0hi = *(const h8*)(pr0 + 16);
        v16h a0 = __builtin_shufflevector(a0lo, a0hi,
            0,1,2,3,4,5,6,7,8,9,10,11,12,13,14,15);
        const _Float16* pr1 = &pbuf[16 + mrow][kt * 32 + ksel];
        h8 a1lo = *(const h8*)pr1;
        h8 a1hi = *(const h8*)(pr1 + 16);
        v16h a1 = __builtin_shufflevector(a1lo, a1hi,
            0,1,2,3,4,5,6,7,8,9,10,11,12,13,14,15);

#pragma unroll
        for (int j = 0; j < 3; ++j) {
          // B fragment: lane holds column col[j], K chunks {0-7,16-23}/{8-15,24-31}
          // along E row col[j] (no transpose load needed).
          const _Float16* er = Eg + (size_t)col[j] * KP + kt * 32 + kbse;
          h8 blo = *(const h8*)er;
          h8 bhi = *(const h8*)(er + 16);
          v16h bf = __builtin_shufflevector(blo, bhi,
              0,1,2,3,4,5,6,7,8,9,10,11,12,13,14,15);

          acc[j][0] = __builtin_amdgcn_wmma_f32_16x16x32_f16(
              false, a0, false, bf, (short)0, acc[j][0], false, false);
          acc[j][1] = __builtin_amdgcn_wmma_f32_16x16x32_f16(
              false, a1, false, bf, (short)0, acc[j][1], false, false);
        }
      }

      // Fuse: alpha_new = inputs[b,t,col] + amax[b] + log(acc)
#pragma unroll
      for (int j = 0; j < 3; ++j) {
        if (col[j] < NC) {
#pragma unroll
          for (int r = 0; r < 8; ++r) {
            int b0 = r + rofs;
            int b1 = 16 + b0;
            float s0 = amax_s[b0] + __logf(acc[j][0][r]);
            float s1 = amax_s[b1] + __logf(acc[j][1][r]);
            alpha_s[b0][col[j]] = inp[((size_t)b0 * T_STEPS + t) * NC + col[j]] + s0;
            alpha_s[b1][col[j]] = inp[((size_t)b1 * T_STEPS + t) * NC + col[j]] + s1;
          }
        }
      }
      if (t + 1 < T_STEPS && col[0] < NC) {
        __builtin_prefetch(&inp[((size_t)rofs * T_STEPS + t + 1) * NC + col[0]], 0, 0);
        __builtin_prefetch(&inp[((size_t)(16 + rofs) * T_STEPS + t + 1) * NC + col[0]], 0, 0);
      }
    }
    __syncthreads();

    // Per-batch max over alpha (16 threads per batch row)
    float pmax = NEGV;
    for (int i = g; i < NC; i += 16) pmax = fmaxf(pmax, alpha_s[rb][i]);
    red_s[rb][g] = pmax;
    __syncthreads();
    if (g == 0) {
      float m = red_s[rb][0];
#pragma unroll
      for (int j = 1; j < 16; ++j) m = fmaxf(m, red_s[rb][j]);
      amax_s[rb] = m;
    }
    __syncthreads();

    // p = exp(alpha - amax) in f16; zero the K padding
    for (int e = tid; e < BATCH * KP; e += 512) {
      int b = e / KP, k = e - b * KP;
      float pv = (k < NC) ? __expf(alpha_s[b][k] - amax_s[b]) : 0.0f;
      pbuf[b][k] = (_Float16)pv;
    }
    __syncthreads();
  }

  // fcc[b] = logsumexp_i(alphaT[b,i]) = amax[b] + log(sum exp(alpha - amax))
  float ps = 0.0f;
  for (int i = g; i < NC; i += 16) ps += __expf(alpha_s[rb][i] - amax_s[rb]);
  red_s[rb][g] = ps;
  __syncthreads();
  if (g == 0) {
    float ssum = 0.0f;
#pragma unroll
    for (int j = 0; j < 16; ++j) ssum += red_s[rb][j];
    fcc[rb] = amax_s[rb] + __logf(ssum);
  }
}

// ---------------------------------------------------------------------------
// Kernel 3: FAL forced-alignment scan. One block per batch, one thread per
// of the 101 lattice states; double-buffered beta in LDS.
// ---------------------------------------------------------------------------
__global__ void asg_fal_kernel(const float* __restrict__ inp,
                               const float* __restrict__ trans,
                               const int* __restrict__ tgtraw,
                               float* __restrict__ fal) {
  const int b = blockIdx.x;
  const int s = threadIdx.x;
  __shared__ int tgts[LL];
  __shared__ float beta[2][LL];

  if (s < LL) {
    // processed targets: odd slots = target+NUM_REPLABELS, else GARBAGE_IDX
    tgts[s] = (s & 1) ? (tgtraw[b * LT + (s >> 1)] + 1) : (NC - 1);
  }
  __syncthreads();

  float self_w = 0.0f, move_w = NEGV;
  int c = 0;
  if (s < LL) {
    c = tgts[s];
    self_w = trans[(1 + c) * NC + c];
    if (s > 0) move_w = trans[(1 + c) * NC + tgts[s - 1]];
    beta[0][s] = (s == 0) ? (inp[(size_t)b * T_STEPS * NC + c] + trans[c]) : NEGV;
  }

  for (int t = 1; t < T_STEPS; ++t) {
    __syncthreads();
    const int prev = (t - 1) & 1, cur = t & 1;
    if (s < LL) {
      float x = beta[prev][s] + self_w;
      float y = ((s > 0) ? beta[prev][s - 1] : NEGV) + move_w;
      float mx = fmaxf(x, y);
      float lae = mx + __logf(__expf(x - mx) + __expf(y - mx));
      beta[cur][s] = inp[((size_t)b * T_STEPS + t) * NC + c] + lae;
    }
  }
  __syncthreads();
  if (s == LL - 1) fal[b] = beta[(T_STEPS - 1) & 1][LL - 1];
}

// ---------------------------------------------------------------------------
// Kernel 4: out = mean_b (fcc[b]-fal[b]) / L
// ---------------------------------------------------------------------------
__global__ void asg_finalize(const float* __restrict__ fcc,
                             const float* __restrict__ fal,
                             float* __restrict__ out) {
  __shared__ float buf[BATCH];
  int t = threadIdx.x;
  buf[t] = (fcc[t] - fal[t]) * (1.0f / (float)LL);
  __syncthreads();
  for (int sh = 16; sh > 0; sh >>= 1) {
    if (t < sh) buf[t] += buf[t + sh];
    __syncthreads();
  }
  if (t == 0) out[0] = buf[0] * (1.0f / (float)BATCH);
}

extern "C" void kernel_launch(void* const* d_in, const int* in_sizes, int n_in,
                              void* d_out, int out_size, void* d_ws, size_t ws_size,
                              hipStream_t stream) {
  const float* inp   = (const float*)d_in[0];   // (32,2000,514) f32
  const float* trans = (const float*)d_in[1];   // (515,514)     f32
  const int*   tgt   = (const int*)d_in[2];     // (32,50)       i32
  float* out = (float*)d_out;

  _Float16* Eg = (_Float16*)d_ws;                                // 544*544*2 B
  float* fcc = (float*)((char*)d_ws + (size_t)NRE * KP * sizeof(_Float16));
  float* fal = fcc + BATCH;

  const int nE = NRE * KP;
  asg_build_expT<<<(nE + 255) / 256, 256, 0, stream>>>(trans, Eg);
  asg_fal_kernel<<<BATCH, 128, 0, stream>>>(inp, trans, tgt, fal);
  asg_fcc_kernel<<<1, 512, 0, stream>>>(inp, trans, Eg, fcc);
  asg_finalize<<<1, BATCH, 0, stream>>>(fcc, fal, out);
}
